// SHINE_3539053052706
// MI455X (gfx1250) — matrix operand order repaired and checked by reading, not compiled, not run on hardware
//
#include <hip/hip_runtime.h>

typedef __bf16 bf16;
typedef __attribute__((ext_vector_type(16))) __bf16 v16bf;
typedef __attribute__((ext_vector_type(8)))  __bf16 v8bf;
typedef __attribute__((ext_vector_type(8)))  float  v8f;

__device__ __forceinline__ float gelu_f(float v) {
    return 0.5f * v * (1.0f + erff(v * 0.70710678118654752f));
}

__global__ void shine_zero(bf16* __restrict__ p, int n)
{
    const int i = blockIdx.x * blockDim.x + threadIdx.x;
    if (i < n) p[i] = (bf16)0.0f;
}

// ---------------------------------------------------------------------------
// Weight pre-pack: fp32 (OC,IC,taps) -> bf16 fragment-ordered
//   pk[(((ocT*taps + t)*(IC/32) + ci)*32 + lane)*16 + e]
// so an A fragment (this code's contiguous k = 16*hi + e order) is one contiguous 32B load.
// ---------------------------------------------------------------------------
__global__ void shine_pack_w(const float* __restrict__ w, bf16* __restrict__ pk,
                             int total, int IC, int taps)
{
    const int idx = blockIdx.x * blockDim.x + threadIdx.x;
    if (idx >= total) return;
    const int nChunk = IC / 32;
    const int e    = idx & 15;
    const int lane = (idx >> 4) & 31;
    int r = idx >> 9;
    const int ci = r % nChunk;  r /= nChunk;
    const int t  = r % taps;
    const int ocT = r / taps;
    const int hi = lane >> 4, lh = lane & 15;
    const int oc = ocT * 16 + lh;
    const int k  = 16 * hi + e;   // A-fragment K mapping
    const int cin = ci * 32 + k;
    pk[idx] = (bf16)w[(oc * IC + cin) * taps + t];
}

// ---------------------------------------------------------------------------
// Implicit-GEMM conv, NHWC bf16 activations, WMMA bf16 -> f32.
// One wave per block: (16*OCB) out-channels x (16*NB) pixels.
// Out-of-bounds taps read a zeroed row (per-lane pointer select, once per tap)
// so every B load is an unconditional 32B global_load_b128 pair.
// Per 32-K chunk: NB B loads + OCB A loads -> OCB*NB WMMAs.
// taps: 9 = dense 3x3, 8 = donut 3x3, 1 = 1x1 GEMM. Epilogue: +skip, gelu,
// one v8bf (16B) store per accumulator half.
// ---------------------------------------------------------------------------
template <int OCB, int NB>
__global__ void __launch_bounds__(32)
shine_wmma_conv(const bf16* __restrict__ in, const bf16* __restrict__ wpk,
                const bf16* __restrict__ skip, bf16* __restrict__ out,
                const bf16* __restrict__ zrow,
                int H, int W, int IC, int ICT, int OCT, int ocOff,
                int taps, int donut, int dil)
{
    const int lane = threadIdx.x & 31;
    const int lh   = lane & 15;
    const int hi   = lane >> 4;
    const int HW   = H * W;
    const int nChunk = IC / 32;

    const int ocT0 = blockIdx.y * OCB;
    int p[NB], y[NB], x[NB];
#pragma unroll
    for (int v = 0; v < NB; ++v) {
        p[v] = blockIdx.x * (16 * NB) + 16 * v + lh;   // flat pixel (n*HW + y*W + x)
        const int rem = p[v] % HW;
        y[v] = rem / W;
        x[v] = rem - y[v] * W;
    }

    v8f acc[OCB][NB];
#pragma unroll
    for (int j = 0; j < OCB; ++j)
#pragma unroll
        for (int v = 0; v < NB; ++v) acc[j][v] = (v8f){};

    for (int t = 0; t < taps; ++t) {
        const int kp = (taps == 1) ? 4 : (donut ? (t < 4 ? t : t + 1) : t);
        const int dy = (kp / 3 - 1) * dil;
        const int dx = (kp % 3 - 1) * dil;
        const bf16* brow[NB];
#pragma unroll
        for (int v = 0; v < NB; ++v) {
            const int iy = y[v] + dy, ix = x[v] + dx;
            const bool ok = (iy >= 0) && (iy < H) && (ix >= 0) && (ix < W);
            // per-lane pointer select: OOB lanes read the zero row
            brow[v] = ok ? (in + (p[v] + dy * W + dx) * ICT + 16 * hi)
                         : (zrow + 16 * hi);
        }
        __builtin_prefetch(brow[0], 0, 0);             // global_prefetch_b8
        const bf16* arow = wpk + ((ocT0 * taps + t) * nChunk) * 512 + lane * 16;

        for (int ci = 0; ci < nChunk; ++ci) {
            v16bf b[NB];
#pragma unroll
            for (int v = 0; v < NB; ++v)
                b[v] = *(const v16bf*)(brow[v] + ci * 32);   // unconditional 32B load
#pragma unroll
            for (int j = 0; j < OCB; ++j) {
                const v16bf a = *(const v16bf*)(arow + (j * taps * nChunk + ci) * 512);
#pragma unroll
                for (int v = 0; v < NB; ++v)
                    acc[j][v] = __builtin_amdgcn_wmma_f32_16x16x32_bf16(
                        false, a, false, b[v], (short)0, acc[j][v], false, false);
            }
        }
    }

    // D: VGPR r holds M = r + 8*hi, N = lh -> 8 consecutive channels, 16B store
#pragma unroll
    for (int j = 0; j < OCB; ++j)
#pragma unroll
    for (int v = 0; v < NB; ++v) {
        const int obase = p[v] * OCT + ocOff + (ocT0 + j) * 16 + 8 * hi;
        v8f a = acc[j][v];
        if (skip) {
            const v8bf s = *(const v8bf*)(skip + obase);  // skip: OCT==64, ocOff==0
#pragma unroll
            for (int r = 0; r < 8; ++r) a[r] += (float)s[r];
        }
        v8bf o;
#pragma unroll
        for (int r = 0; r < 8; ++r) o[r] = (bf16)gelu_f(a[r]);
        *(v8bf*)(out + obase) = o;
    }
}

// ---------------------------------------------------------------------------
// Stem: img NCHW fp32 (2,5,H,W). base NHWC bf16 (n,y,x,64):
//   ch 0..47 = gelu(conv0b 1x1 on img ch2), ch 48..63 = gelu(conv0a 3x3 on {0,1,3,4})
// ---------------------------------------------------------------------------
__global__ void shine_stem(const float* __restrict__ img, const float* __restrict__ w0a,
                           const float* __restrict__ w0b, bf16* __restrict__ base,
                           int Nn, int H, int W)
{
    const int idx = blockIdx.x * blockDim.x + threadIdx.x;
    const int HW = H * W;
    if (idx >= Nn * HW) return;
    const int n = idx / HW, rem = idx - n * HW;
    const int y = rem / W, x = rem - y * W;
    const float* im = img + (long)n * 5 * HW;
    bf16* bo = base + idx * 64;

    const float tv = im[2 * HW + rem];
    for (int o = 0; o < 48; ++o)
        bo[o] = (bf16)gelu_f(w0b[o] * tv);

    const int srcc[4] = {0, 1, 3, 4};
    for (int o = 0; o < 16; ++o) {
        float acc = 0.f;
        for (int ic = 0; ic < 4; ++ic) {
            const float* ch = im + srcc[ic] * HW;
            for (int ky = 0; ky < 3; ++ky) {
                const int iy = y + ky - 1;
                if (iy < 0 || iy >= H) continue;
                for (int kx = 0; kx < 3; ++kx) {
                    const int ix = x + kx - 1;
                    if (ix < 0 || ix >= W) continue;
                    acc += w0a[((o * 4 + ic) * 3 + ky) * 3 + kx] * ch[iy * W + ix];
                }
            }
        }
        bo[48 + o] = (bf16)gelu_f(acc);
    }
}

// NHWC 2x2 mean; consecutive threads cover C=64 -> coalesced
__global__ void shine_avgpool(const bf16* __restrict__ in, bf16* __restrict__ out,
                              int total, int Ho, int Wo)
{
    const int idx = blockIdx.x * blockDim.x + threadIdx.x;
    if (idx >= total) return;
    const int c = idx & 63;
    int pp = idx >> 6;
    const int x = pp % Wo; pp /= Wo;
    const int y = pp % Ho;
    const int n = pp / Ho;
    const int Wi = Wo * 2;
    const bf16* s = in + (((n * (Ho * 2) + 2 * y) * Wi + 2 * x) * 64) + c;
    const float v = ((float)s[0] + (float)s[64] + (float)s[Wi * 64] + (float)s[Wi * 64 + 64]) * 0.25f;
    out[idx] = (bf16)v;
}

// Bilinear, half-pixel centers (matches jax.image.resize 'bilinear'), NHWC
__global__ void shine_upsample(const bf16* __restrict__ in, bf16* __restrict__ out,
                               int total, int Hs, int Ws, int Ho, int Wo)
{
    const int idx = blockIdx.x * blockDim.x + threadIdx.x;
    if (idx >= total) return;
    const int c = idx & 63;
    int pp = idx >> 6;
    const int x = pp % Wo; pp /= Wo;
    const int y = pp % Ho;
    const int n = pp / Ho;
    float sy = (y + 0.5f) * ((float)Hs / (float)Ho) - 0.5f;
    float sx = (x + 0.5f) * ((float)Ws / (float)Wo) - 0.5f;
    sy = fminf(fmaxf(sy, 0.f), (float)(Hs - 1));
    sx = fminf(fmaxf(sx, 0.f), (float)(Ws - 1));
    const int y0 = (int)sy, x0 = (int)sx;
    const int y1 = min(y0 + 1, Hs - 1), x1 = min(x0 + 1, Ws - 1);
    const float fy = sy - y0, fx = sx - x0;
    const bf16* s = in + (long)n * Hs * Ws * 64 + c;
    const float v00 = (float)s[(y0 * Ws + x0) * 64], v01 = (float)s[(y0 * Ws + x1) * 64];
    const float v10 = (float)s[(y1 * Ws + x0) * 64], v11 = (float)s[(y1 * Ws + x1) * 64];
    const float v = v00 * (1.f - fy) * (1.f - fx) + v01 * (1.f - fy) * fx
                  + v10 * fy * (1.f - fx) + v11 * fy * fx;
    out[idx] = (bf16)v;
}

// 64 -> 1 final 1x1 conv, fp32 NCHW output (N,1,H,W)
__global__ void shine_lastconv(const bf16* __restrict__ in, const float* __restrict__ lw,
                               float* __restrict__ out, int total)
{
    const int idx = blockIdx.x * blockDim.x + threadIdx.x;
    if (idx >= total) return;
    const bf16* s = in + idx * 64;
    float acc = 0.f;
    for (int c = 0; c < 64; ++c) acc += lw[c] * (float)s[c];
    out[idx] = acc;
}

// ---------------------------------------------------------------------------
extern "C" void kernel_launch(void* const* d_in, const int* in_sizes, int n_in,
                              void* d_out, int out_size, void* d_ws, size_t ws_size,
                              hipStream_t stream)
{
    (void)in_sizes; (void)n_in; (void)out_size; (void)ws_size;

    const float* img     = (const float*)d_in[0];
    const float* w0a     = (const float*)d_in[1];
    const float* w0b     = (const float*)d_in[2];
    const float* first_p = (const float*)d_in[3];
    const float* cl10    = (const float*)d_in[4];   // (4,2,64,64,3,3)
    const float* cl2     = (const float*)d_in[5];
    const float* dilp    = (const float*)d_in[6];   // (4,64,64,8)
    const float* dilres  = (const float*)d_in[7];
    const float* upsw    = (const float*)d_in[8];   // (4,64,64,1,1)
    const float* out1w   = (const float*)d_in[9];   // (288,576)
    const float* out2w   = (const float*)d_in[10];  // (64,288)
    const float* lastw   = (const float*)d_in[11];  // (1,64)

    const int Nn = 2, H0 = 256, W0 = 256;
    const int RB = 64 * 64 * 9;
    const int DP = 64 * 64 * 8;
    const int UW = 64 * 64;

    char* ws = (char*)d_ws;
    size_t off = 0;
    auto alloc = [&](size_t bytes) -> void* {
        void* pp = ws + off;
        off = (off + bytes + 255) & ~(size_t)255;
        return pp;
    };
    bf16* base0 = (bf16*)alloc((size_t)Nn * 64  * 256 * 256 * 2);
    bf16* base1 = (bf16*)alloc((size_t)Nn * 64  * 128 * 128 * 2);
    bf16* base2 = (bf16*)alloc((size_t)Nn * 64  *  64 *  64 * 2);
    bf16* base3 = (bf16*)alloc((size_t)Nn * 64  *  32 *  32 * 2);
    bf16* tres  = (bf16*)alloc((size_t)Nn * 64  * 256 * 256 * 2);
    bf16* mbuf  = (bf16*)alloc((size_t)Nn * 64  * 256 * 256 * 2);
    bf16* ubuf  = (bf16*)alloc((size_t)Nn * 64  * 256 * 256 * 2);
    bf16* cat   = (bf16*)alloc((size_t)Nn * 576 * 256 * 256 * 2);
    bf16* b288  = (bf16*)alloc((size_t)Nn * 288 * 256 * 256 * 2);
    bf16* b64   = (bf16*)alloc((size_t)Nn * 64  * 256 * 256 * 2);
    bf16* zrow  = (bf16*)alloc((size_t)1024 * 2);

    shine_zero<<<4, 256, 0, stream>>>(zrow, 1024);

    // ---- pack all weights to bf16 fragment order (L2-resident, ~2.2MB) ----
    auto pack = [&](const float* w, int OC, int IC, int taps) -> bf16* {
        const int total = OC * IC * taps;
        bf16* pk = (bf16*)alloc((size_t)total * 2);
        shine_pack_w<<<(total + 255) / 256, 256, 0, stream>>>(w, pk, total, IC, taps);
        return pk;
    };
    bf16* pk_first = pack(first_p, 64, 64, 8);
    bf16* pk_cl10[8];
    bf16* pk_cl2[8];
    for (int i = 0; i < 8; ++i) {
        pk_cl10[i] = pack(cl10 + i * RB, 64, 64, 9);
        pk_cl2[i]  = pack(cl2  + i * RB, 64, 64, 9);
    }
    bf16* pk_dilp[4];
    bf16* pk_dilres[4];
    bf16* pk_ups[4];
    for (int i = 0; i < 4; ++i) {
        pk_dilp[i]   = pack(dilp   + i * DP, 64, 64, 8);
        pk_dilres[i] = pack(dilres + i * DP, 64, 64, 8);
        pk_ups[i]    = pack(upsw   + i * UW, 64, 64, 1);
    }
    bf16* pk_out1 = pack(out1w, 288, 576, 1);
    bf16* pk_out2 = pack(out2w, 64, 288, 1);

    // ---- conv launchers: M=64(or 32) x N=32 per wave ----
    auto conv = [&](const bf16* in, const bf16* wpk, const bf16* skip, bf16* out,
                    int H, int W, int IC, int ICT, int OC, int OCT, int ocOff,
                    int taps, int donut, int dil) {
        if (OC == 64) {
            dim3 g((unsigned)(Nn * H * W / 32), 1);
            shine_wmma_conv<4, 2><<<g, 32, 0, stream>>>(in, wpk, skip, out, zrow, H, W,
                                                        IC, ICT, OCT, ocOff, taps, donut, dil);
        } else { // OC == 288
            dim3 g((unsigned)(Nn * H * W / 32), (unsigned)(OC / 32));
            shine_wmma_conv<2, 2><<<g, 32, 0, stream>>>(in, wpk, skip, out, zrow, H, W,
                                                        IC, ICT, OCT, ocOff, taps, donut, dil);
        }
    };
    auto resblock = [&](bf16* base, bf16* w1, bf16* w2, int H, int W) {
        conv(base, w1, nullptr, tres, H, W, 64, 64, 64, 64, 0, 9, 0, 1);
        conv(tres, w2, base,    base, H, W, 64, 64, 64, 64, 0, 9, 0, 1);
    };
    auto entry = [&](bf16* dw, int dil, bf16* uw, int level, int ocOff, bf16* baseL) {
        const int Hl = H0 >> level, Wl = W0 >> level;
        conv(baseL, dw, nullptr, mbuf, Hl, Wl, 64, 64, 64, 64, 0, 8, 1, dil);
        const bf16* src = mbuf;
        if (level > 0) {
            const int total = Nn * 64 * H0 * W0;
            shine_upsample<<<(total + 255) / 256, 256, 0, stream>>>(mbuf, ubuf, total, Hl, Wl, H0, W0);
            src = ubuf;
        }
        conv(src, uw, nullptr, cat, H0, W0, 64, 64, 64, 576, ocOff, 1, 0, 1);
    };
    auto pool = [&](const bf16* in, bf16* out, int Ho) {
        const int total = Nn * 64 * Ho * Ho;
        shine_avgpool<<<(total + 255) / 256, 256, 0, stream>>>(in, out, total, Ho, Ho);
    };

    // ---- stem ----
    {
        const int total = Nn * H0 * W0;
        shine_stem<<<(total + 255) / 256, 256, 0, stream>>>(img, w0a, w0b, base0, Nn, H0, W0);
    }
    // ---- level 0 (256) ----
    conv(base0, pk_first, nullptr, cat, 256, 256, 64, 64, 64, 576, 0, 8, 1, 1);      // outs[0]
    resblock(base0, pk_cl10[0], pk_cl10[1], 256, 256);
    conv(base0, pk_dilres[0], nullptr, cat, 256, 256, 64, 64, 64, 576, 64, 8, 1, 3); // outs[1]
    resblock(base0, pk_cl2[0], pk_cl2[1], 256, 256);
    entry(pk_dilp[0], 5, pk_ups[0], 0, 128, base0);                                   // outs[2]
    // ---- level 1 (128) ----
    pool(base0, base1, 128);
    resblock(base1, pk_cl10[2], pk_cl10[3], 128, 128);
    entry(pk_dilres[1], 6, pk_ups[1], 1, 192, base1);                                 // outs[3]
    resblock(base1, pk_cl2[2], pk_cl2[3], 128, 128);
    entry(pk_dilp[1], 8, pk_ups[1], 1, 256, base1);                                   // outs[4]
    // ---- level 2 (64) ----
    pool(base1, base2, 64);
    resblock(base2, pk_cl10[4], pk_cl10[5], 64, 64);
    entry(pk_dilres[2], 8, pk_ups[2], 2, 320, base2);                                 // outs[5]
    resblock(base2, pk_cl2[4], pk_cl2[5], 64, 64);
    entry(pk_dilp[2], 10, pk_ups[2], 2, 384, base2);                                  // outs[6]
    // ---- level 3 (32) ----
    pool(base2, base3, 32);
    resblock(base3, pk_cl10[6], pk_cl10[7], 32, 32);
    entry(pk_dilres[3], 9, pk_ups[3], 3, 448, base3);                                 // outs[7]
    resblock(base3, pk_cl2[6], pk_cl2[7], 32, 32);
    entry(pk_dilp[3], 11, pk_ups[3], 3, 512, base3);                                  // outs[8]
    // ---- head ----
    conv(cat,  pk_out1, nullptr, b288, 256, 256, 576, 576, 288, 288, 0, 1, 0, 1);
    conv(b288, pk_out2, nullptr, b64,  256, 256, 288, 288, 64,  64,  0, 1, 0, 1);
    {
        const int total = Nn * H0 * W0;
        shine_lastconv<<<(total + 255) / 256, 256, 0, stream>>>(b64, lastw, (float*)d_out, total);
    }
}
